// GNN_Model_85074712199805
// MI455X (gfx1250) — matrix operand order repaired and checked by reading, not compiled, not run on hardware
//
#include <hip/hip_runtime.h>
#include <hip/hip_bf16.h>
#include <math.h>

// ---------------------------------------------------------------------------
// GNN forward for MI455X (gfx1250). Dense node-feature GEMMs run on
// v_wmma_f32_16x16x32_bf16 with BOTH operands pre-packed in workspace into
// a contiguous-k WMMA tile layout (k = 16*(lane/16) + j, wave32): the GEMM inner
// loop is pure b128 vector loads + WMMA with a compile-time-unrolled subtile
// count (no divergent-looking branches -> no exec save/restore, accumulators
// stay pinned). Edge-indexed segment ops (GAT / transformer attention,
// softmax, pooling) use fp32 VALU with native float atomics; average degree
// ~4 so atomic contention is negligible and intermediates are L2-resident.
// ---------------------------------------------------------------------------

typedef __attribute__((ext_vector_type(16))) __bf16         v16bf;
typedef __attribute__((ext_vector_type(16))) unsigned short v16us;
typedef __attribute__((ext_vector_type(8)))  float          v8f;

constexpr int NN  = 50000;   // nodes
constexpr int EE  = 200000;  // edges
constexpr int E2E = EE + NN; // edges + self loops
constexpr int NG  = 64;      // graphs

__device__ __forceinline__ unsigned short f2bf(float f) {
  unsigned u = __float_as_uint(f);
  u += 0x7fffu + ((u >> 16) & 1u);          // round-to-nearest-even
  return (unsigned short)(u >> 16);
}
__device__ __forceinline__ float eluf(float x) {
  return x > 0.f ? x : (__expf(x) - 1.f);
}
// order-preserving float <-> uint map for atomicMax-based segment max
__device__ __forceinline__ unsigned encf(float f) {
  unsigned u = __float_as_uint(f);
  return (u & 0x80000000u) ? ~u : (u | 0x80000000u);
}
__device__ __forceinline__ float decf(unsigned k) {
  unsigned u = (k & 0x80000000u) ? (k & 0x7fffffffu) : ~k;
  return __uint_as_float(u);
}
__device__ __forceinline__ long long gtid() {
  return (long long)blockIdx.x * blockDim.x + threadIdx.x;
}

// ------------------------------ utility kernels ----------------------------

__global__ void zero_f32(float* __restrict__ p, long long n) {
  long long t = gtid();
  if (t < n) p[t] = 0.f;
}

// ---------------------- WMMA operand packing -------------------------------
// A pack: for tile (tM, kt), lane l (row = l&15, hi = l>>4) the 16 bf16 are
//   j in 0..15 -> A[tM*16+row][kt*32 + hi*16 + j]
// stored contiguously so the GEMM loads one 32B vector per lane per K-step.
__global__ void pack_a_kernel(const float* __restrict__ A,
                              unsigned short* __restrict__ out,
                              int M, int K) {
  long long t = gtid();
  int kt32 = K >> 5;
  long long ntile32 = (long long)(M >> 4) * kt32 * 32;
  if (t >= ntile32) return;
  int lane = (int)(t & 31);
  long long tile = t >> 5;
  int kt = (int)(tile % kt32);
  int tM = (int)(tile / kt32);
  int l15 = lane & 15, hi = lane >> 4;
  const float* arow = A + (size_t)(tM * 16 + l15) * K + kt * 32 + hi * 16;
  float4 a0 = *(const float4*)(arow);
  float4 a1 = *(const float4*)(arow + 4);
  float4 b0 = *(const float4*)(arow + 8);
  float4 b1 = *(const float4*)(arow + 12);
  v16us o;
  o[0]  = f2bf(a0.x); o[1]  = f2bf(a0.y); o[2]  = f2bf(a0.z); o[3]  = f2bf(a0.w);
  o[4]  = f2bf(a1.x); o[5]  = f2bf(a1.y); o[6]  = f2bf(a1.z); o[7]  = f2bf(a1.w);
  o[8]  = f2bf(b0.x); o[9]  = f2bf(b0.y); o[10] = f2bf(b0.z); o[11] = f2bf(b0.w);
  o[12] = f2bf(b1.x); o[13] = f2bf(b1.y); o[14] = f2bf(b1.z); o[15] = f2bf(b1.w);
  *(v16us*)(out + t * 16) = o;
}

// B pack (weights, once): tile (kt, nt), lane l (col = l&15, hi = l>>4):
//   j in 0..15 -> B[kt*32 + hi*16 + j][nt*16 + col]
__global__ void pack_b_kernel(const float* __restrict__ B,
                              unsigned short* __restrict__ out,
                              int K, int N) {
  long long t = gtid();
  int nt16 = N >> 4;
  long long ntile32 = (long long)(K >> 5) * nt16 * 32;
  if (t >= ntile32) return;
  int lane = (int)(t & 31);
  long long tile = t >> 5;
  int nt = (int)(tile % nt16);
  int kt = (int)(tile / nt16);
  int col = nt * 16 + (lane & 15);
  int hi  = lane >> 4;
  const float* bp = B + (size_t)(kt * 32 + hi * 16) * N + col;
  v16us o;
#pragma unroll
  for (int j = 0; j < 16; ++j) o[j] = f2bf(bp[(size_t)j * N]);
  *(v16us*)(out + t * 16) = o;
}

// ------------------------------ WMMA GEMM ----------------------------------
// D[M,N] = act( A[M,K] @ B[K,N] + bias + addend ); A,B pre-packed bf16 tiles.
// One wave computes a 16 x (NT*16) tile; NT is compile-time so the subtile
// loop fully unrolls with no conditionals (EXEC constant, accs stay pinned).
template <int NT>
__global__ __launch_bounds__(256) void wmma_gemm_bf16(
    const unsigned short* __restrict__ Apk,
    const unsigned short* __restrict__ Bpk,
    const float* __restrict__ bias, const float* __restrict__ addend,
    float* __restrict__ D, int M, int N, int K, int act, int tilesN) {
  int wave  = blockIdx.x * (blockDim.x >> 5) + (threadIdx.x >> 5);
  int lane  = threadIdx.x & 31;
  int tileM = wave / tilesN;
  int tn    = wave - tileM * tilesN;
  if (tileM * 16 >= M) return;            // uniform per wave
  int l15 = lane & 15;
  int hi  = lane >> 4;
  int kt32 = K >> 5;
  int nt16 = N >> 4;

  v8f acc[NT];
#pragma unroll
  for (int i = 0; i < NT; ++i)
#pragma unroll
    for (int r = 0; r < 8; ++r) acc[i][r] = 0.f;

  const unsigned short* ap = Apk + ((size_t)tileM * kt32 * 32 + lane) * 16;
  const unsigned short* bp = Bpk + ((size_t)(tn * NT) * 32 + lane) * 16;
  size_t bstride = (size_t)nt16 * 512;     // ushorts per K-step of B tiles
  for (int kt = 0; kt < kt32; ++kt) {
    __builtin_prefetch(ap + 512, 0, 1);    // speculative: next A K-block
    v16us au = *(const v16us*)ap;
    v16bf a  = __builtin_bit_cast(v16bf, au);
#pragma unroll
    for (int nt = 0; nt < NT; ++nt) {
      v16us bu = *(const v16us*)(bp + nt * 512);
      v16bf b  = __builtin_bit_cast(v16bf, bu);
      acc[nt] = __builtin_amdgcn_wmma_f32_16x16x32_bf16(
          false, a, false, b, (short)0, acc[nt], false, false);
    }
    ap += 512;
    bp += bstride;
  }
  // C layout: vgpr r -> row r (lanes<16) / r+8 (lanes>=16), col = lane&15
#pragma unroll
  for (int nt = 0; nt < NT; ++nt) {
    int col = (tn * NT + nt) * 16 + l15;
#pragma unroll
    for (int r = 0; r < 8; ++r) {
      int mrow   = tileM * 16 + hi * 8 + r;
      size_t idx = (size_t)mrow * N + col;
      float v = acc[nt][r];
      if (bias)   v += bias[col];
      if (addend) v += addend[idx];
      if (act)    v = eluf(v);
      D[idx] = v;
    }
  }
}

// ------------------------------ graph kernels ------------------------------

__global__ void edge_deg_kernel(const int* __restrict__ ei,
                                const float* __restrict__ ea,
                                float* __restrict__ deg,
                                float* __restrict__ easum, int E) {
  long long e = gtid();
  if (e >= E) return;
  int dst = ei[E + e];
  atomicAdd(&deg[dst], 1.f);
  atomicAdd(&easum[dst], ea[e]);
}

__global__ void lealoop_kernel(const float* __restrict__ deg,
                               const float* __restrict__ easum,
                               float* __restrict__ leal, int Nn) {
  long long v = gtid();
  if (v < Nn) leal[v] = easum[v] / fmaxf(deg[v], 1.f);
}

// agg[dst] += ea * x[src]   (IN_CH = 32, 4 channels per thread)
__global__ void edge_agg_kernel(const int* __restrict__ ei,
                                const float* __restrict__ ea,
                                const float* __restrict__ x,
                                float* __restrict__ agg, int E) {
  long long t = gtid();
  if (t >= (long long)E * 8) return;
  int e = (int)(t >> 3);
  int c = (int)(t & 7) << 2;
  int src = ei[e], dst = ei[E + e];
  float w = ea[e];
  float4 xv = *(const float4*)(x + (size_t)src * 32 + c);
  float* o  = agg + (size_t)dst * 32 + c;
  atomicAdd(o + 0, w * xv.x);
  atomicAdd(o + 1, w * xv.y);
  atomicAdd(o + 2, w * xv.z);
  atomicAdd(o + 3, w * xv.w);
}

// a_src[n,h] = sum_c h[n,h,c]*as[h,c] ; a_dst likewise
__global__ void att_coef_kernel(const float* __restrict__ HP,
                                const float* __restrict__ as_,
                                const float* __restrict__ ad_,
                                float* __restrict__ asrc,
                                float* __restrict__ adst,
                                int Nn, int H, int C) {
  long long t = gtid();
  if (t >= (long long)Nn * H) return;
  int n = (int)(t / H), h = (int)(t - (long long)n * H);
  const float* hp = HP + ((size_t)n * H + h) * C;
  const float* av = as_ + h * C;
  const float* dv = ad_ + h * C;
  float sa = 0.f, sd = 0.f;
  for (int c = 0; c < C; c += 4) {
    float4 hv = *(const float4*)(hp + c);
    float4 a4 = *(const float4*)(av + c);
    float4 d4 = *(const float4*)(dv + c);
    sa += hv.x * a4.x + hv.y * a4.y + hv.z * a4.z + hv.w * a4.w;
    sd += hv.x * d4.x + hv.y * d4.y + hv.z * d4.z + hv.w * d4.w;
  }
  asrc[t] = sa;
  adst[t] = sd;
}

// hcoef[h] = sum_c we[h*C+c]*ae[h*C+c]
__global__ void hcoef_kernel(const float* __restrict__ we,
                             const float* __restrict__ ae,
                             float* __restrict__ hc, int H, int C) {
  int h = threadIdx.x;
  if (h >= H) return;
  float s = 0.f;
  for (int c = 0; c < C; ++c) s += we[h * C + c] * ae[h * C + c];
  hc[h] = s;
}

__global__ void init_softmax_kernel(unsigned* __restrict__ maxb,
                                    float* __restrict__ sumb, long long n) {
  long long t = gtid();
  if (t >= n) return;
  maxb[t] = 0x007fffffu;   // encf(-inf)
  sumb[t] = 0.f;
}

// GAT logits over E2 edges (self loops appended): leaky_relu + segment max
__global__ void gat_logits_kernel(const int* __restrict__ ei,
                                  const float* __restrict__ ea,
                                  const float* __restrict__ leal,
                                  const float* __restrict__ asrc,
                                  const float* __restrict__ adst,
                                  const float* __restrict__ hc,
                                  float* __restrict__ logb,
                                  unsigned* __restrict__ maxb,
                                  int E, int Etot, int H) {
  long long t = gtid();
  if (t >= (long long)Etot * H) return;
  int e = (int)(t / H), h = (int)(t - (long long)e * H);
  int src, dst;
  float lea;
  if (e < E) { src = ei[e]; dst = ei[E + e]; lea = ea[e]; }
  else       { src = dst = e - E;            lea = leal[src]; }
  float a = asrc[(size_t)src * H + h] + adst[(size_t)dst * H + h] + lea * hc[h];
  a = a > 0.f ? a : 0.2f * a;             // leaky_relu(0.2)
  logb[t] = a;
  atomicMax(&maxb[(size_t)dst * H + h], encf(a));
}

// transformer logits over original E edges: (q[dst].(k[src]+ea*we))/sqrt(C)
__global__ void trans_logits_kernel(const int* __restrict__ ei,
                                    const float* __restrict__ ea,
                                    const float* __restrict__ Q,
                                    const float* __restrict__ Kf,
                                    const float* __restrict__ we,
                                    float* __restrict__ logb,
                                    unsigned* __restrict__ maxb,
                                    int E, int H, int C) {
  long long t = gtid();
  if (t >= (long long)E * H) return;
  int e = (int)(t / H), h = (int)(t - (long long)e * H);
  int src = ei[e], dst = ei[E + e];
  float eav = ea[e];
  const float* q = Q  + ((size_t)dst * H + h) * C;
  const float* k = Kf + ((size_t)src * H + h) * C;
  const float* w = we + h * C;
  float s = 0.f;
  for (int c = 0; c < C; c += 4) {
    float4 qv = *(const float4*)(q + c);
    float4 kv = *(const float4*)(k + c);
    float4 wv = *(const float4*)(w + c);
    s += qv.x * (kv.x + eav * wv.x) + qv.y * (kv.y + eav * wv.y) +
         qv.z * (kv.z + eav * wv.z) + qv.w * (kv.w + eav * wv.w);
  }
  s *= rsqrtf((float)C);
  logb[t] = s;
  atomicMax(&maxb[(size_t)dst * H + h], encf(s));
}

// w = exp(logit - max[dst]) ; segment sum of w
__global__ void softmax_exp_kernel(const int* __restrict__ ei,
                                   float* __restrict__ logb,
                                   const unsigned* __restrict__ maxb,
                                   float* __restrict__ sumb,
                                   int E, int Etot, int H) {
  long long t = gtid();
  if (t >= (long long)Etot * H) return;
  int e = (int)(t / H), h = (int)(t - (long long)e * H);
  int dst = (e < E) ? ei[E + e] : (e - E);
  float m = decf(maxb[(size_t)dst * H + h]);
  if (!(m > -INFINITY)) m = 0.f;          // isolated dst: max stays -inf -> 0
  float w = __expf(logb[t] - m);
  logb[t] = w;
  atomicAdd(&sumb[(size_t)dst * H + h], w);
}

// out[dst,h,:] += h[src,h,:] * w/(sum[dst,h]+eps)   (4 channels per thread)
__global__ void gat_accum_kernel(const int* __restrict__ ei,
                                 const float* __restrict__ HP,
                                 const float* __restrict__ W,
                                 const float* __restrict__ sumb,
                                 float* __restrict__ out,
                                 int E, int Etot, int H, int C) {
  long long t   = gtid();
  int c4cnt     = C >> 2;
  long long per = (long long)H * c4cnt;
  if (t >= (long long)Etot * per) return;
  int e = (int)(t / per);
  int r = (int)(t - (long long)e * per);
  int h = r / c4cnt;
  int c = (r - h * c4cnt) << 2;
  int src, dst;
  if (e < E) { src = ei[e]; dst = ei[E + e]; } else { src = dst = e - E; }
  float coeff = W[(size_t)e * H + h] / (sumb[(size_t)dst * H + h] + 1e-16f);
  float4 hv = *(const float4*)(HP + ((size_t)src * H + h) * C + c);
  float* o  = out + ((size_t)dst * H + h) * C + c;
  atomicAdd(o + 0, hv.x * coeff);
  atomicAdd(o + 1, hv.y * coeff);
  atomicAdd(o + 2, hv.z * coeff);
  atomicAdd(o + 3, hv.w * coeff);
}

// out[dst,h,:] += (v[src,h,:] + ea*we[h,:]) * w/(sum+eps)
__global__ void trans_accum_kernel(const int* __restrict__ ei,
                                   const float* __restrict__ ea,
                                   const float* __restrict__ V,
                                   const float* __restrict__ we,
                                   const float* __restrict__ W,
                                   const float* __restrict__ sumb,
                                   float* __restrict__ out,
                                   int E, int H, int C) {
  long long t   = gtid();
  int c4cnt     = C >> 2;
  long long per = (long long)H * c4cnt;
  if (t >= (long long)E * per) return;
  int e = (int)(t / per);
  int r = (int)(t - (long long)e * per);
  int h = r / c4cnt;
  int c = (r - h * c4cnt) << 2;
  int src = ei[e], dst = ei[E + e];
  float eav   = ea[e];
  float coeff = W[(size_t)e * H + h] / (sumb[(size_t)dst * H + h] + 1e-16f);
  float4 vv = *(const float4*)(V + ((size_t)src * H + h) * C + c);
  float4 wv = *(const float4*)(we + h * C + c);
  float* o  = out + ((size_t)dst * H + h) * C + c;
  atomicAdd(o + 0, (vv.x + eav * wv.x) * coeff);
  atomicAdd(o + 1, (vv.y + eav * wv.y) * coeff);
  atomicAdd(o + 2, (vv.z + eav * wv.z) * coeff);
  atomicAdd(o + 3, (vv.w + eav * wv.w) * coeff);
}

__global__ void epilogue_concat_kernel(const float* __restrict__ s,
                                       const float* __restrict__ b,
                                       float* __restrict__ d,
                                       long long n, int HC) {
  long long t = gtid();
  if (t < n) d[t] = eluf(s[t] + b[(int)(t % HC)]);
}

__global__ void epilogue_mean_kernel(const float* __restrict__ s,
                                     const float* __restrict__ b,
                                     float* __restrict__ d,
                                     int Nn, int H, int C) {
  long long t = gtid();
  if (t >= (long long)Nn * C) return;
  int n = (int)(t / C), c = (int)(t - (long long)n * C);
  float acc = 0.f;
  for (int h = 0; h < H; ++h) acc += s[((size_t)n * H + h) * C + c];
  d[t] = eluf(acc / (float)H + b[c]);
}

__global__ void add_elu_kernel(const float* __restrict__ a,
                               const float* __restrict__ b,
                               float* __restrict__ d, long long n) {
  long long t = gtid();
  if (t < n) d[t] = eluf(a[t] + b[t]);
}

__global__ void pool_kernel(const float* __restrict__ h,
                            const int* __restrict__ batch,
                            float* __restrict__ g, int Nn, int C) {
  long long t = gtid();
  if (t >= (long long)Nn * C) return;
  int n = (int)(t / C), c = (int)(t - (long long)n * C);
  atomicAdd(&g[(size_t)batch[n] * C + c], h[t]);
}

// tiny dense layers (M = 64 graphs)
__global__ void fc_kernel(const float* __restrict__ A,
                          const float* __restrict__ W,
                          const float* __restrict__ bias,
                          float* __restrict__ D,
                          int M, int N, int K, int act) {
  int t = blockIdx.x * blockDim.x + threadIdx.x;
  if (t >= M * N) return;
  int m = t / N, n = t - m * N;
  float s = bias[n];
  for (int k = 0; k < K; ++k) s += A[(size_t)m * K + k] * W[(size_t)k * N + n];
  if (act) s = eluf(s);
  D[t] = s;
}

// ------------------------------ host driver --------------------------------

extern "C" void kernel_launch(void* const* d_in, const int* in_sizes, int n_in,
                              void* d_out, int out_size, void* d_ws,
                              size_t ws_size, hipStream_t stream) {
  (void)in_sizes; (void)n_in; (void)out_size; (void)ws_size;
  // pytree order: x, edge_attr, params(dict insertion order), edge_index,
  // batch, n_graphs
  const float* x        = (const float*)d_in[0];
  const float* ea       = (const float*)d_in[1];
  const float* gc_wrel  = (const float*)d_in[2];
  const float* gc_brel  = (const float*)d_in[3];
  const float* gc_wroot = (const float*)d_in[4];
  const float* g2_w  = (const float*)d_in[5];
  const float* g2_as = (const float*)d_in[6];
  const float* g2_ad = (const float*)d_in[7];
  const float* g2_we = (const float*)d_in[8];
  const float* g2_ae = (const float*)d_in[9];
  const float* g2_b  = (const float*)d_in[10];
  const float* g3_w  = (const float*)d_in[11];
  const float* g3_as = (const float*)d_in[12];
  const float* g3_ad = (const float*)d_in[13];
  const float* g3_we = (const float*)d_in[14];
  const float* g3_ae = (const float*)d_in[15];
  const float* g3_b  = (const float*)d_in[16];
  const float* t_wq  = (const float*)d_in[17];
  const float* t_bq  = (const float*)d_in[18];
  const float* t_wk  = (const float*)d_in[19];
  const float* t_bk  = (const float*)d_in[20];
  const float* t_wv  = (const float*)d_in[21];
  const float* t_bv  = (const float*)d_in[22];
  const float* t_we  = (const float*)d_in[23];
  const float* t_ws  = (const float*)d_in[24];
  const float* t_bs  = (const float*)d_in[25];
  const float* g4_w  = (const float*)d_in[26];
  const float* g4_as = (const float*)d_in[27];
  const float* g4_ad = (const float*)d_in[28];
  const float* g4_we = (const float*)d_in[29];
  const float* g4_ae = (const float*)d_in[30];
  const float* g4_b  = (const float*)d_in[31];
  const float* fc1_w = (const float*)d_in[32];
  const float* fc1_b = (const float*)d_in[33];
  const float* fco_w = (const float*)d_in[34];
  const float* fco_b = (const float*)d_in[35];
  const int* eidx  = (const int*)d_in[36];
  const int* batch = (const int*)d_in[37];
  float* out = (float*)d_out;

  // ---- workspace: packed-bf16 region (ushort) then f32 buffers ----
  unsigned short* wb = (unsigned short*)d_ws;
  size_t wo = 0;
  auto walloc = [&](size_t n) { size_t r = wo; wo += (n + 127) & ~127ull; return r; };
  size_t BPK_GCREL  = walloc(32 * 32);
  size_t BPK_GCROOT = walloc(32 * 32);
  size_t BPK_G2W = walloc(32 * 256);
  size_t BPK_G3W = walloc(256 * 512);
  size_t BPK_TWQ = walloc(128 * 512);
  size_t BPK_TWK = walloc(128 * 512);
  size_t BPK_TWV = walloc(128 * 512);
  size_t BPK_TWS = walloc(128 * 512);
  size_t BPK_G4W = walloc(512 * 256);
  size_t APK     = walloc((size_t)NN * 512);       // packed activations (max K=512)
  float* fbase = (float*)((char*)d_ws + ((wo * 2 + 255) & ~255ull));
  size_t fo = 0;
  auto falloc = [&](size_t n) { size_t r = fo; fo += (n + 63) & ~63ull; return r; };
  float*    DEG   = fbase + falloc(NN);
  float*    EASUM = fbase + falloc(NN);
  float*    LEAL  = fbase + falloc(NN);
  float*    HC    = fbase + falloc(64);
  float*    ASRC  = fbase + falloc((size_t)NN * 4);
  float*    ADST  = fbase + falloc((size_t)NN * 4);
  float*    SUMB  = fbase + falloc((size_t)NN * 4);
  unsigned* MAXB  = (unsigned*)(fbase + falloc((size_t)NN * 4));
  float*    LOGW  = fbase + falloc((size_t)E2E * 4);
  float*    AGG   = fbase + falloc((size_t)NN * 32);
  float*    H1    = fbase + falloc((size_t)NN * 32);
  float*    H3    = fbase + falloc((size_t)NN * 128);
  float*    MED0  = fbase + falloc((size_t)NN * 256);   // H2 / H5
  float*    BIG0  = fbase + falloc((size_t)NN * 512);   // projections / Q
  float*    BIG1  = fbase + falloc((size_t)NN * 512);   // segment-sum out / tmp
  float*    BIG2  = fbase + falloc((size_t)NN * 512);   // K, later H4
  float*    BIG3  = fbase + falloc((size_t)NN * 512);   // V
  float*    BIG4  = fbase + falloc((size_t)NN * 512);   // skip
  float*    GP    = fbase + falloc((size_t)NG * 256);
  float*    GH    = fbase + falloc((size_t)NG * 64);

  auto nb = [](long long n, int bs) { return (unsigned)((n + bs - 1) / bs); };
  auto packB = [&](const float* W, size_t off, int K, int N) {
    long long th = (long long)(K / 32) * (N / 16) * 32;
    pack_b_kernel<<<nb(th, 256), 256, 0, stream>>>(W, wb + off, K, N);
  };
  auto packA = [&](const float* A, int K) {
    long long th = (long long)(NN / 16) * (K / 32) * 32;
    pack_a_kernel<<<nb(th, 256), 256, 0, stream>>>(A, wb + APK, NN, K);
  };
  auto gemm = [&](size_t Boff, const float* bias, const float* addend,
                  float* D, int Nc, int Kc, int act) {
    if (Nc == 32) {                       // NT=2, one 16x32 tile per wave
      int tilesN = 1;
      long long waves = (long long)(NN / 16) * tilesN;
      wmma_gemm_bf16<2><<<nb(waves, 8), 256, 0, stream>>>(
          wb + APK, wb + Boff, bias, addend, D, NN, Nc, Kc, act, tilesN);
    } else {                              // NT=4, 16x64 tiles per wave
      int tilesN = Nc / 64;
      long long waves = (long long)(NN / 16) * tilesN;
      wmma_gemm_bf16<4><<<nb(waves, 8), 256, 0, stream>>>(
          wb + APK, wb + Boff, bias, addend, D, NN, Nc, Kc, act, tilesN);
    }
  };
  auto run_gat = [&](const float* HPp, const float* as_, const float* ad_,
                     const float* we_, const float* ae_, int H, int C) {
    att_coef_kernel<<<nb((long long)NN * H, 256), 256, 0, stream>>>(
        HPp, as_, ad_, ASRC, ADST, NN, H, C);
    hcoef_kernel<<<1, 32, 0, stream>>>(we_, ae_, HC, H, C);
    init_softmax_kernel<<<nb((long long)NN * H, 256), 256, 0, stream>>>(
        MAXB, SUMB, (long long)NN * H);
    gat_logits_kernel<<<nb((long long)E2E * H, 256), 256, 0, stream>>>(
        eidx, ea, LEAL, ASRC, ADST, HC, LOGW, MAXB, EE, E2E, H);
    softmax_exp_kernel<<<nb((long long)E2E * H, 256), 256, 0, stream>>>(
        eidx, LOGW, MAXB, SUMB, EE, E2E, H);
    zero_f32<<<nb((long long)NN * H * C, 256), 256, 0, stream>>>(
        BIG1, (long long)NN * H * C);
    gat_accum_kernel<<<nb((long long)E2E * H * (C / 4), 256), 256, 0, stream>>>(
        eidx, HPp, LOGW, SUMB, BIG1, EE, E2E, H, C);
  };

  // ---- pack weights to WMMA B-tile bf16 layout (once) ----
  packB(gc_wrel, BPK_GCREL, 32, 32);   packB(gc_wroot, BPK_GCROOT, 32, 32);
  packB(g2_w, BPK_G2W, 32, 256);       packB(g3_w, BPK_G3W, 256, 512);
  packB(t_wq, BPK_TWQ, 128, 512);      packB(t_wk, BPK_TWK, 128, 512);
  packB(t_wv, BPK_TWV, 128, 512);      packB(t_ws, BPK_TWS, 128, 512);
  packB(g4_w, BPK_G4W, 512, 256);

  // ---- self loops: deg, sum(ea), loop attr ----
  zero_f32<<<nb(NN, 256), 256, 0, stream>>>(DEG, NN);
  zero_f32<<<nb(NN, 256), 256, 0, stream>>>(EASUM, NN);
  edge_deg_kernel<<<nb(EE, 256), 256, 0, stream>>>(eidx, ea, DEG, EASUM, EE);
  lealoop_kernel<<<nb(NN, 256), 256, 0, stream>>>(DEG, EASUM, LEAL, NN);

  // ---- gc: agg = segsum(ea*x[src]); h1 = elu(agg@Wrel + b + x@Wroot) ----
  zero_f32<<<nb((long long)NN * 32, 256), 256, 0, stream>>>(AGG, (long long)NN * 32);
  edge_agg_kernel<<<nb((long long)EE * 8, 256), 256, 0, stream>>>(eidx, ea, x, AGG, EE);
  packA(AGG, 32);
  gemm(BPK_GCREL, gc_brel, nullptr, BIG1, 32, 32, 0);
  packA(x, 32);
  gemm(BPK_GCROOT, nullptr, BIG1, H1, 32, 32, 1);

  // ---- g2: GAT H=4 C=64, concat -> H2 [N,256] ----
  packA(H1, 32);
  gemm(BPK_G2W, nullptr, nullptr, BIG0, 256, 32, 0);
  run_gat(BIG0, g2_as, g2_ad, g2_we, g2_ae, 4, 64);
  epilogue_concat_kernel<<<nb((long long)NN * 256, 256), 256, 0, stream>>>(
      BIG1, g2_b, MED0, (long long)NN * 256, 256);

  // ---- g3: GAT H=4 C=128, mean -> H3 [N,128] ----
  packA(MED0, 256);
  gemm(BPK_G3W, nullptr, nullptr, BIG0, 512, 256, 0);
  run_gat(BIG0, g3_as, g3_ad, g3_we, g3_ae, 4, 128);
  epilogue_mean_kernel<<<nb((long long)NN * 128, 256), 256, 0, stream>>>(
      BIG1, g3_b, H3, NN, 4, 128);

  // ---- transformer: H=4 C=128 over original edges -> H4 [N,512] ----
  packA(H3, 128);                                  // one packed A, 4 GEMMs
  gemm(BPK_TWQ, t_bq, nullptr, BIG0, 512, 128, 0); // Q
  gemm(BPK_TWK, t_bk, nullptr, BIG2, 512, 128, 0); // K
  gemm(BPK_TWV, t_bv, nullptr, BIG3, 512, 128, 0); // V
  gemm(BPK_TWS, t_bs, nullptr, BIG4, 512, 128, 0); // skip
  init_softmax_kernel<<<nb((long long)NN * 4, 256), 256, 0, stream>>>(
      MAXB, SUMB, (long long)NN * 4);
  trans_logits_kernel<<<nb((long long)EE * 4, 256), 256, 0, stream>>>(
      eidx, ea, BIG0, BIG2, t_we, LOGW, MAXB, EE, 4, 128);
  softmax_exp_kernel<<<nb((long long)EE * 4, 256), 256, 0, stream>>>(
      eidx, LOGW, MAXB, SUMB, EE, EE, 4);
  zero_f32<<<nb((long long)NN * 512, 256), 256, 0, stream>>>(BIG1, (long long)NN * 512);
  trans_accum_kernel<<<nb((long long)EE * 4 * 32, 256), 256, 0, stream>>>(
      eidx, ea, BIG3, t_we, LOGW, SUMB, BIG1, EE, 4, 128);
  add_elu_kernel<<<nb((long long)NN * 512, 256), 256, 0, stream>>>(
      BIG1, BIG4, BIG2, (long long)NN * 512);      // H4 -> BIG2

  // ---- g4: GAT H=4 C=64, concat -> H5 [N,256] ----
  packA(BIG2, 512);
  gemm(BPK_G4W, nullptr, nullptr, BIG0, 256, 512, 0);
  run_gat(BIG0, g4_as, g4_ad, g4_we, g4_ae, 4, 64);
  epilogue_concat_kernel<<<nb((long long)NN * 256, 256), 256, 0, stream>>>(
      BIG1, g4_b, MED0, (long long)NN * 256, 256);

  // ---- graph pooling + FC head ----
  zero_f32<<<nb((long long)NG * 256, 256), 256, 0, stream>>>(GP, (long long)NG * 256);
  pool_kernel<<<nb((long long)NN * 256, 256), 256, 0, stream>>>(MED0, batch, GP, NN, 256);
  fc_kernel<<<nb(NG * 64, 256), 256, 0, stream>>>(GP, fc1_w, fc1_b, GH, NG, 64, 256, 1);
  fc_kernel<<<nb(NG * 10, 256), 256, 0, stream>>>(GH, fco_w, fco_b, out, NG, 10, 64, 0);
}
